// DotProductAttention_67422396612827
// MI455X (gfx1250) — matrix operand order repaired and checked by reading, not compiled, not run on hardware
//
#include <hip/hip_runtime.h>
#include <math.h>

typedef __attribute__((ext_vector_type(16))) _Float16 v16h;
typedef __attribute__((ext_vector_type(8)))  _Float16 v8h;
typedef __attribute__((ext_vector_type(4)))  _Float16 v4h;
typedef __attribute__((ext_vector_type(8)))  float    v8f;
typedef __attribute__((ext_vector_type(4)))  int      v4i;

#define B_   16
#define S_   2048
#define D_   128
#define TQ   128      // q rows per block (8 waves x 16)
#define TK   64       // keys per tile
#define NW   8        // waves per block
#define NT   (S_ / TK)
#define KPAD 136      // halfs per K-LDS row   (272 B: 16B aligned, 4-bank step)
#define VTP  72       // halfs per V^T-LDS row (144 B: 16B aligned, 36-bank step)

// ---- CDNA5 async global->LDS (ASYNCcnt) with portable fallback ----
#if __has_builtin(__builtin_amdgcn_global_load_async_to_lds_b128)
#define HAVE_ASYNC 1
typedef __attribute__((address_space(1))) v4i gv4i;
typedef __attribute__((address_space(3))) v4i lv4i;
#define ASYNC_B128(g, l)                                                   \
    __builtin_amdgcn_global_load_async_to_lds_b128(                        \
        (gv4i*)(g), (lv4i*)(l), 0, 0)
#else
#define HAVE_ASYNC 0
#define ASYNC_B128(g, l) (*(uint4*)(l) = *(const uint4*)(g))
#endif

static __device__ __forceinline__ void wait_async() {
#if HAVE_ASYNC
#if __has_builtin(__builtin_amdgcn_s_wait_asynccnt)
    __builtin_amdgcn_s_wait_asynccnt(0);
#else
    asm volatile("s_wait_asynccnt 0" ::: "memory");
#endif
#endif
}

// =====================================================================
// Pre-pass: K -> f16 row-major [b][s][d]; V -> f16 transposed [b][d][s]
// =====================================================================
__launch_bounds__(256, 1)
__global__ void cvt_kv_kernel(const float* __restrict__ K,
                              const float* __restrict__ V,
                              _Float16* __restrict__ Kh,
                              _Float16* __restrict__ VtG)
{
    __shared__ _Float16 Vt[D_ * VTP];
    const int tid = threadIdx.x;
    const int blk = blockIdx.x;
    const int b   = blk >> 5;     // / NT
    const int kt  = blk & 31;

    const float4* kg = (const float4*)(K + ((size_t)b * S_ + kt * TK) * D_);
    const float4* vg = (const float4*)(V + ((size_t)b * S_ + kt * TK) * D_);
    _Float16* khb = Kh + ((size_t)b * S_ + kt * TK) * D_;

    #pragma unroll
    for (int i = 0; i < 8; ++i) {
        int f   = tid + 256 * i;       // 2048 float4 per 64x128 tile
        int row = f >> 5;
        int col = (f & 31) << 2;
        float4 kv = kg[f];
        v4h hk = { (_Float16)kv.x, (_Float16)kv.y, (_Float16)kv.z, (_Float16)kv.w };
        *(v4h*)&khb[row * D_ + col] = hk;
        float4 vv = vg[f];
        Vt[(col + 0) * VTP + row] = (_Float16)vv.x;
        Vt[(col + 1) * VTP + row] = (_Float16)vv.y;
        Vt[(col + 2) * VTP + row] = (_Float16)vv.z;
        Vt[(col + 3) * VTP + row] = (_Float16)vv.w;
    }
    __syncthreads();
    // write transposed tile out coalesced: VtG[b][d][s]
    #pragma unroll
    for (int i = 0; i < 4; ++i) {
        int f   = tid + 256 * i;       // 1024 16B chunks (256 rows x 4)
        int row = f >> 2;
        int c   = f & 3;
        uint4 val = *(const uint4*)&Vt[row * VTP + c * 8];
        *(uint4*)&VtG[((size_t)(b * D_ + row)) * S_ + kt * TK + c * 8] = val;
    }
}

// =====================================================================
// Main flash-attention kernel: async double-buffered f16 tiles + WMMA
// =====================================================================
__launch_bounds__(256, 1)
__global__ void fa_fwd_f16_kernel(const float* __restrict__ Q,
                                  const _Float16* __restrict__ Kh,
                                  const _Float16* __restrict__ VtG,
                                  float* __restrict__ O)
{
    __shared__ _Float16 Kl[2][TK * KPAD];     // K tile, [key][d]
    __shared__ _Float16 Vt[2][D_ * VTP];      // V tile transposed, [d][key]
    __shared__ _Float16 Pl[NW * 16 * TK];     // per-wave P relayout scratch

    const int tid  = threadIdx.x;
    const int wave = tid >> 5;
    const int lane = tid & 31;
    const int l15  = lane & 15;
    const int hi   = (lane >> 4) & 1;

    const int blk   = blockIdx.x;
    const int b     = blk >> 4;
    const int qb    = blk & 15;
    const int qrow0 = qb * TQ + wave * 16;

    // 1/sqrt(128) * log2(e): softmax in exp2 domain (native v_exp_f32)
    const float scale = 0.08838834764831845f * 1.44269504088896340f;

    // ---- Q A-fragments (16x32 f16 layout), pre-scaled ----
    v16h qf[4];
    {
        const float* qrow = Q + ((size_t)b * S_ + qrow0 + l15) * D_;
        #pragma unroll
        for (int c = 0; c < 4; ++c) {
            #pragma unroll
            for (int e = 0; e < 16; ++e) {
                int k = (e < 8 ? e : e + 8) + (hi ? 8 : 0);
                qf[c][e] = (_Float16)(qrow[32 * c + k] * scale);
            }
        }
    }

    v16h ones;
    #pragma unroll
    for (int e = 0; e < 16; ++e) ones[e] = (_Float16)1.0f;

    v8f acc[8];
    #pragma unroll
    for (int s = 0; s < 8; ++s) acc[s] = (v8f){};
    float m[8], l[8];
    #pragma unroll
    for (int r = 0; r < 8; ++r) { m[r] = -INFINITY; l[r] = 0.0f; }

    _Float16* pw = &Pl[wave * 16 * TK];

    const _Float16* kbase = Kh  + (size_t)b * S_ * D_;
    const _Float16* vbase = VtG + (size_t)b * D_ * S_;

    // issue async fills of buffer `buf` with tile `kt` (8 b128 per thread)
    auto issue = [&](int buf, int kt) {
        const _Float16* kb = kbase + (size_t)kt * TK * D_;
        const _Float16* vb = vbase + kt * TK;
        #pragma unroll
        for (int i = 0; i < 4; ++i) {
            int f = tid + 256 * i;              // 1024 chunks each
            int kr = f >> 4, kc = f & 15;       // K: 64 rows x 16 chunks
            ASYNC_B128(kb + kr * D_ + kc * 8, &Kl[buf][kr * KPAD + kc * 8]);
            int vr = f >> 3, vc = f & 7;        // V^T: 128 rows x 8 chunks
            ASYNC_B128(vb + (size_t)vr * S_ + vc * 8, &Vt[buf][vr * VTP + vc * 8]);
        }
    };

    issue(0, 0);

    for (int kt = 0; kt < NT; ++kt) {
        const int cur = kt & 1;
        wait_async();        // own async writes to buf[cur] landed
        __syncthreads();     // everyone's landed; buf[cur^1] fully drained
        if (kt + 1 < NT) issue(cur ^ 1, kt + 1);   // overlaps compute below

        const _Float16* Kc = Kl[cur];
        const _Float16* Vc = Vt[cur];

        // ---- S = Q K^T : four 16x16 f32 fragments ----
        v8f sf[4];
        #pragma unroll
        for (int j = 0; j < 4; ++j) {
            v8f c = (v8f){};
            #pragma unroll
            for (int cc = 0; cc < 4; ++cc) {
                const _Float16* src = &Kc[(16 * j + l15) * KPAD + 32 * cc + (hi ? 8 : 0)];
                v8h lo = *(const v8h*)src;
                v8h hg = *(const v8h*)(src + 16);
                v16h bf;
                #pragma unroll
                for (int e = 0; e < 8; ++e) { bf[e] = lo[e]; bf[e + 8] = hg[e]; }
                c = __builtin_amdgcn_wmma_f32_16x16x32_f16(
                        false, qf[cc], false, bf, (short)0, c, false, false);
            }
            sf[j] = c;
        }

        // ---- online softmax: row max + correction (exp2 domain) ----
        float corr[8];
        #pragma unroll
        for (int r = 0; r < 8; ++r) {
            float t = fmaxf(fmaxf(sf[0][r], sf[1][r]), fmaxf(sf[2][r], sf[3][r]));
            #pragma unroll
            for (int x = 8; x >= 1; x >>= 1)
                t = fmaxf(t, __shfl_xor(t, x, 32));
            float mn = fmaxf(m[r], t);
            corr[r] = __builtin_exp2f(m[r] - mn);
            m[r] = mn;
        }
        // ---- P = exp2(S - m) -> per-wave LDS scratch (C layout) ----
        #pragma unroll
        for (int r = 0; r < 8; ++r) {
            int row = r + 8 * hi;
            _Float16* pr = &pw[row * TK + l15];
            #pragma unroll
            for (int j = 0; j < 4; ++j)
                pr[16 * j] = (_Float16)__builtin_exp2f(sf[j][r] - m[r]);
        }

        // ---- P back as two A-fragments (same-wave LDS ops in order) ----
        v16h pf0, pf1;
        {
            const _Float16* base = &pw[l15 * TK];
            v8h x0 = *(const v8h*)(base + (hi ? 8 : 0));
            v8h x1 = *(const v8h*)(base + (hi ? 24 : 16));
            v8h y0 = *(const v8h*)(base + 32 + (hi ? 8 : 0));
            v8h y1 = *(const v8h*)(base + 32 + (hi ? 24 : 16));
            #pragma unroll
            for (int e = 0; e < 8; ++e) {
                pf0[e] = x0[e]; pf0[e + 8] = x1[e];
                pf1[e] = y0[e]; pf1[e + 8] = y1[e];
            }
        }

        // ---- row sums via ones-trick: l = l*corr + P·1 ----
        {
            v8f rs = (v8f){};
            rs = __builtin_amdgcn_wmma_f32_16x16x32_f16(
                     false, pf0, false, ones, (short)0, rs, false, false);
            rs = __builtin_amdgcn_wmma_f32_16x16x32_f16(
                     false, pf1, false, ones, (short)0, rs, false, false);
            #pragma unroll
            for (int r = 0; r < 8; ++r) l[r] = l[r] * corr[r] + rs[r];
        }

        // ---- acc = acc*corr + P V ----
        #pragma unroll
        for (int s = 0; s < 8; ++s) {
            v8f c;
            #pragma unroll
            for (int r = 0; r < 8; ++r) c[r] = acc[s][r] * corr[r];
            const _Float16* src0 = &Vc[(16 * s + l15) * VTP + (hi ? 8 : 0)];
            v8h lo0 = *(const v8h*)src0;
            v8h hg0 = *(const v8h*)(src0 + 16);
            v16h bf0;
            #pragma unroll
            for (int e = 0; e < 8; ++e) { bf0[e] = lo0[e]; bf0[e + 8] = hg0[e]; }
            c = __builtin_amdgcn_wmma_f32_16x16x32_f16(
                    false, pf0, false, bf0, (short)0, c, false, false);
            const _Float16* src1 = src0 + 32;
            v8h lo1 = *(const v8h*)src1;
            v8h hg1 = *(const v8h*)(src1 + 16);
            v16h bf1;
            #pragma unroll
            for (int e = 0; e < 8; ++e) { bf1[e] = lo1[e]; bf1[e + 8] = hg1[e]; }
            acc[s] = __builtin_amdgcn_wmma_f32_16x16x32_f16(
                         false, pf1, false, bf1, (short)0, c, false, false);
        }
    }

    // ---- epilogue: O = acc / l ----
    float rl[8];
    #pragma unroll
    for (int r = 0; r < 8; ++r) rl[r] = 1.0f / l[r];
    float* orow = O + ((size_t)b * S_ + qrow0) * D_;
    #pragma unroll
    for (int s = 0; s < 8; ++s) {
        #pragma unroll
        for (int r = 0; r < 8; ++r)
            orow[(size_t)(r + 8 * hi) * D_ + 16 * s + l15] = acc[s][r] * rl[r];
    }
}

// =====================================================================
// Fallback (ws too small): round-2 style all-in-one kernel, f32 inputs
// =====================================================================
__launch_bounds__(256, 1)
__global__ void fa_fwd_f32_kernel(const float* __restrict__ Q,
                                  const float* __restrict__ K,
                                  const float* __restrict__ V,
                                  float* __restrict__ O)
{
    __shared__ _Float16 Kl[TK * KPAD];
    __shared__ _Float16 Vt[D_ * VTP];
    __shared__ _Float16 Pl[NW * 16 * TK];

    const int tid  = threadIdx.x;
    const int wave = tid >> 5;
    const int lane = tid & 31;
    const int l15  = lane & 15;
    const int hi   = (lane >> 4) & 1;

    const int blk   = blockIdx.x;
    const int b     = blk >> 4;
    const int qb    = blk & 15;
    const int qrow0 = qb * TQ + wave * 16;

    const float scale = 0.08838834764831845f * 1.44269504088896340f;

    v16h qf[4];
    {
        const float* qrow = Q + ((size_t)b * S_ + qrow0 + l15) * D_;
        #pragma unroll
        for (int c = 0; c < 4; ++c)
            #pragma unroll
            for (int e = 0; e < 16; ++e) {
                int k = (e < 8 ? e : e + 8) + (hi ? 8 : 0);
                qf[c][e] = (_Float16)(qrow[32 * c + k] * scale);
            }
    }
    v16h ones;
    #pragma unroll
    for (int e = 0; e < 16; ++e) ones[e] = (_Float16)1.0f;

    v8f acc[8];
    #pragma unroll
    for (int s = 0; s < 8; ++s) acc[s] = (v8f){};
    float m[8], l[8];
    #pragma unroll
    for (int r = 0; r < 8; ++r) { m[r] = -INFINITY; l[r] = 0.0f; }

    _Float16* pw = &Pl[wave * 16 * TK];
    const float4* kg = (const float4*)(K + (size_t)b * S_ * D_);
    const float4* vg = (const float4*)(V + (size_t)b * S_ * D_);

    for (int kt = 0; kt < NT; ++kt) {
        __syncthreads();
        #pragma unroll
        for (int i = 0; i < 8; ++i) {
            int f   = tid + 256 * i;
            int row = f >> 5;
            int col = (f & 31) << 2;
            float4 kv = kg[(size_t)kt * TK * D_ / 4 + f];
            _Float16* kd = &Kl[row * KPAD + col];
            kd[0] = (_Float16)kv.x; kd[1] = (_Float16)kv.y;
            kd[2] = (_Float16)kv.z; kd[3] = (_Float16)kv.w;
            float4 vv = vg[(size_t)kt * TK * D_ / 4 + f];
            Vt[(col + 0) * VTP + row] = (_Float16)vv.x;
            Vt[(col + 1) * VTP + row] = (_Float16)vv.y;
            Vt[(col + 2) * VTP + row] = (_Float16)vv.z;
            Vt[(col + 3) * VTP + row] = (_Float16)vv.w;
        }
        __syncthreads();

        v8f sf[4];
        #pragma unroll
        for (int j = 0; j < 4; ++j) {
            v8f c = (v8f){};
            #pragma unroll
            for (int cc = 0; cc < 4; ++cc) {
                const _Float16* src = &Kl[(16 * j + l15) * KPAD + 32 * cc + (hi ? 8 : 0)];
                v8h lo = *(const v8h*)src;
                v8h hg = *(const v8h*)(src + 16);
                v16h bf;
                #pragma unroll
                for (int e = 0; e < 8; ++e) { bf[e] = lo[e]; bf[e + 8] = hg[e]; }
                c = __builtin_amdgcn_wmma_f32_16x16x32_f16(
                        false, qf[cc], false, bf, (short)0, c, false, false);
            }
            sf[j] = c;
        }

        float corr[8];
        #pragma unroll
        for (int r = 0; r < 8; ++r) {
            float t = fmaxf(fmaxf(sf[0][r], sf[1][r]), fmaxf(sf[2][r], sf[3][r]));
            #pragma unroll
            for (int x = 8; x >= 1; x >>= 1)
                t = fmaxf(t, __shfl_xor(t, x, 32));
            float mn = fmaxf(m[r], t);
            corr[r] = __builtin_exp2f(m[r] - mn);
            m[r] = mn;
        }
        #pragma unroll
        for (int r = 0; r < 8; ++r) {
            int row = r + 8 * hi;
            _Float16* pr = &pw[row * TK + l15];
            #pragma unroll
            for (int j = 0; j < 4; ++j)
                pr[16 * j] = (_Float16)__builtin_exp2f(sf[j][r] - m[r]);
        }

        v16h pf0, pf1;
        {
            const _Float16* base = &pw[l15 * TK];
            v8h x0 = *(const v8h*)(base + (hi ? 8 : 0));
            v8h x1 = *(const v8h*)(base + (hi ? 24 : 16));
            v8h y0 = *(const v8h*)(base + 32 + (hi ? 8 : 0));
            v8h y1 = *(const v8h*)(base + 32 + (hi ? 24 : 16));
            #pragma unroll
            for (int e = 0; e < 8; ++e) {
                pf0[e] = x0[e]; pf0[e + 8] = x1[e];
                pf1[e] = y0[e]; pf1[e + 8] = y1[e];
            }
        }
        {
            v8f rs = (v8f){};
            rs = __builtin_amdgcn_wmma_f32_16x16x32_f16(
                     false, pf0, false, ones, (short)0, rs, false, false);
            rs = __builtin_amdgcn_wmma_f32_16x16x32_f16(
                     false, pf1, false, ones, (short)0, rs, false, false);
            #pragma unroll
            for (int r = 0; r < 8; ++r) l[r] = l[r] * corr[r] + rs[r];
        }
        #pragma unroll
        for (int s = 0; s < 8; ++s) {
            v8f c;
            #pragma unroll
            for (int r = 0; r < 8; ++r) c[r] = acc[s][r] * corr[r];
            const _Float16* src0 = &Vt[(16 * s + l15) * VTP + (hi ? 8 : 0)];
            v8h lo0 = *(const v8h*)src0;
            v8h hg0 = *(const v8h*)(src0 + 16);
            v16h bf0;
            #pragma unroll
            for (int e = 0; e < 8; ++e) { bf0[e] = lo0[e]; bf0[e + 8] = hg0[e]; }
            c = __builtin_amdgcn_wmma_f32_16x16x32_f16(
                    false, pf0, false, bf0, (short)0, c, false, false);
            const _Float16* src1 = src0 + 32;
            v8h lo1 = *(const v8h*)src1;
            v8h hg1 = *(const v8h*)(src1 + 16);
            v16h bf1;
            #pragma unroll
            for (int e = 0; e < 8; ++e) { bf1[e] = lo1[e]; bf1[e + 8] = hg1[e]; }
            acc[s] = __builtin_amdgcn_wmma_f32_16x16x32_f16(
                         false, pf1, false, bf1, (short)0, c, false, false);
        }
    }

    float rl[8];
    #pragma unroll
    for (int r = 0; r < 8; ++r) rl[r] = 1.0f / l[r];
    float* orow = O + ((size_t)b * S_ + qrow0) * D_;
    #pragma unroll
    for (int s = 0; s < 8; ++s)
        #pragma unroll
        for (int r = 0; r < 8; ++r)
            orow[(size_t)(r + 8 * hi) * D_ + 16 * s + l15] = acc[s][r] * rl[r];
}

extern "C" void kernel_launch(void* const* d_in, const int* in_sizes, int n_in,
                              void* d_out, int out_size, void* d_ws, size_t ws_size,
                              hipStream_t stream) {
    const float* Q = (const float*)d_in[0];
    const float* K = (const float*)d_in[1];
    const float* V = (const float*)d_in[2];
    float* O = (float*)d_out;
    (void)in_sizes; (void)n_in; (void)out_size;

    const size_t need = (size_t)2 * B_ * S_ * D_ * sizeof(_Float16);  // 16.8 MB
    if (d_ws != nullptr && ws_size >= need) {
        _Float16* Kh  = (_Float16*)d_ws;
        _Float16* VtG = Kh + (size_t)B_ * S_ * D_;
        cvt_kv_kernel<<<dim3(B_ * NT), dim3(256), 0, stream>>>(K, V, Kh, VtG);
        fa_fwd_f16_kernel<<<dim3(B_ * (S_ / TQ)), dim3(256), 0, stream>>>(Q, Kh, VtG, O);
    } else {
        fa_fwd_f32_kernel<<<dim3(B_ * (S_ / TQ)), dim3(256), 0, stream>>>(Q, K, V, O);
    }
}
